// MultiHeadAttention_52725018526311
// MI455X (gfx1250) — compile-verified, matrix operand order checked by reading, not run on hardware
//
#include <hip/hip_runtime.h>

typedef _Float16 h8   __attribute__((ext_vector_type(8)));
typedef _Float16 v16h __attribute__((ext_vector_type(16)));
typedef float    v8f  __attribute__((ext_vector_type(8)));

#define B_  2
#define T_  2048
#define C_  1024
#define H_  16
#define D_  64
#define BT_ (B_ * T_)          // 4096
#define N3_ (3 * C_)           // 3072

// ---------------------------------------------------------------------------
// Fragment load: A/B fragment of a 16x16x32 f16 WMMA.
// Per ISA 7.12.2: lane r (=lane&15) owns row `row`; half (=lane>>4) selects the
// K sub-range; halves i<8 are K = kk+8*half.., halves i>=8 are K = kk+8*half+16..
// Both chunks are contiguous 16B in row-major memory -> two b128 loads.
// ---------------------------------------------------------------------------
__device__ __forceinline__ v16h load_frag(const _Float16* __restrict__ base,
                                          int row, int rowStride, int kk, int hf) {
  const _Float16* p = base + (size_t)row * rowStride + kk + 8 * hf;
  h8 lo = *(const h8*)p;
  h8 hi = *(const h8*)(p + 16);
  return __builtin_shufflevector(lo, hi, 0,1,2,3,4,5,6,7,8,9,10,11,12,13,14,15);
}

__device__ __forceinline__ v8f wmma_f16(v16h a, v16h b, v8f c) {
  return __builtin_amdgcn_wmma_f32_16x16x32_f16(false, a, false, b, (short)0, c,
                                                false, false);
}

// ---------------------------------------------------------------------------
// Precision / layout prep kernels
// ---------------------------------------------------------------------------
__global__ void k_f32_to_f16(const float* __restrict__ in, _Float16* __restrict__ out, int n) {
  int i = blockIdx.x * blockDim.x + threadIdx.x;
  if (i < n) out[i] = (_Float16)in[i];
}

// W [rows=K][cols=N] f32 -> Wt [N][K] f16 (coalesced reads)
__global__ void k_transpose_f16(const float* __restrict__ in, _Float16* __restrict__ out,
                                int rows, int cols) {
  int i = blockIdx.x * blockDim.x + threadIdx.x;
  if (i < rows * cols) {
    int k = i / cols, n = i % cols;
    out[(size_t)n * rows + k] = (_Float16)in[i];
  }
}

// ---------------------------------------------------------------------------
// 64x64 wave-tile GEMM core: acc[4][4] += A[64 rows] x Bt[64 rows]^T over K=C_.
// 16 WMMA per 16 b128 loads/lane -> 32 FLOP/byte.
// ---------------------------------------------------------------------------
__device__ __forceinline__ void gemm_tile64(const _Float16* __restrict__ A,
                                            const _Float16* __restrict__ Bt,
                                            int mbase, int nbase,
                                            v8f acc[4][4], int r, int hf) {
  for (int kk = 0; kk < C_; kk += 32) {
    v16h a[4], bfr[4];
    #pragma unroll
    for (int i = 0; i < 4; ++i) a[i]   = load_frag(A,  mbase + 16 * i + r, C_, kk, hf);
    #pragma unroll
    for (int i = 0; i < 4; ++i) bfr[i] = load_frag(Bt, nbase + 16 * i + r, C_, kk, hf);
    #pragma unroll
    for (int mi = 0; mi < 4; ++mi)
      #pragma unroll
      for (int ni = 0; ni < 4; ++ni)
        acc[mi][ni] = wmma_f16(a[mi], bfr[ni], acc[mi][ni]);
  }
}

// ---------------------------------------------------------------------------
// GEMM1: qkv = xh[4096,1024] @ Wqkv  (Wt stored transposed [3072,1024]).
// Block = 4 waves = 128x128 tile. Epilogue scatters f16 into
// Q[B,H,T,D], K[B,H,T,D], Vt[B,H,D,T].
// ---------------------------------------------------------------------------
__global__ __launch_bounds__(128) void k_gemm_qkv(const _Float16* __restrict__ xh,
                                                  const _Float16* __restrict__ wt,
                                                  _Float16* __restrict__ qb,
                                                  _Float16* __restrict__ kb,
                                                  _Float16* __restrict__ vtb) {
  const int lane = threadIdx.x & 31, wave = threadIdx.x >> 5;
  const int r = lane & 15, hf = lane >> 4;
  const int mbase = blockIdx.y * 128 + (wave >> 1) * 64;
  const int nbase = blockIdx.x * 128 + (wave & 1) * 64;

  v8f acc[4][4] = {};
  gemm_tile64(xh, wt, mbase, nbase, acc, r, hf);

  #pragma unroll
  for (int mi = 0; mi < 4; ++mi)
    #pragma unroll
    for (int ni = 0; ni < 4; ++ni)
      #pragma unroll
      for (int j = 0; j < 8; ++j) {
        int m = mbase + 16 * mi + j + 8 * hf;   // row of D-tile (ISA C/D layout)
        int n = nbase + 16 * ni + r;            // col
        int b = m >> 11, t = m & (T_ - 1);
        int which = n >> 10, hd = n & (C_ - 1);
        int h = hd >> 6, d = hd & (D_ - 1);
        int bh = b * H_ + h;
        _Float16 v = (_Float16)acc[mi][ni][j];
        if (which == 0)      qb[(size_t)bh * T_ * D_ + (size_t)t * D_ + d] = v;
        else if (which == 1) kb[(size_t)bh * T_ * D_ + (size_t)t * D_ + d] = v;
        else                 vtb[(size_t)bh * D_ * T_ + (size_t)d * T_ + t] = v;
      }
}

// ---------------------------------------------------------------------------
// Flash attention (causal). One wave = 32 queries (two 16-query tiles sharing
// the K/V fragments) of one (b,h).
// S' = K_blk x Q^T  -> D-layout of S' == A-frag layout of P for P x V.
// ---------------------------------------------------------------------------
__global__ __launch_bounds__(256) void k_attn(const _Float16* __restrict__ qbuf,
                                              const _Float16* __restrict__ kbuf,
                                              const _Float16* __restrict__ vtbuf,
                                              _Float16* __restrict__ attnout) {
  const int lane = threadIdx.x & 31, wave = threadIdx.x >> 5;
  const int r = lane & 15, hf = lane >> 4;
  const int w = blockIdx.x * 8 + wave;          // 2048 waves total
  const int bh = w >> 6;                        // T_/32 = 64 q-tiles of 32 per (b,h)
  const int qt = w & 63;
  const int b = bh >> 4, h = bh & (H_ - 1);
  const int qbase = qt * 32;

  const _Float16* Q  = qbuf  + (size_t)bh * T_ * D_;
  const _Float16* K  = kbuf  + (size_t)bh * T_ * D_;
  const _Float16* Vt = vtbuf + (size_t)bh * D_ * T_;

  // Q fragments (B-operand of S'): [q-subtile][d-chunk], reused every key block
  v16h qf[2][2];
  #pragma unroll
  for (int q2 = 0; q2 < 2; ++q2)
    #pragma unroll
    for (int dc = 0; dc < 2; ++dc)
      qf[q2][dc] = load_frag(Q, qbase + 16 * q2 + r, D_, 32 * dc, hf);

  float m_run[2] = {-3.0e38f, -3.0e38f};
  float l_run[2] = {0.0f, 0.0f};
  v8f o[2][4] = {};
  const float scale = 0.125f;                    // D^-0.5
  const int nb = (qbase >> 5) + 1;               // causal: key blocks of 32

  for (int kbI = 0; kbI < nb; ++kbI) {
    const int kbase = kbI * 32;

    v16h ka[2][2];                               // [key-subtile][d-chunk]
    #pragma unroll
    for (int kti = 0; kti < 2; ++kti)
      #pragma unroll
      for (int dc = 0; dc < 2; ++dc)
        ka[kti][dc] = load_frag(K, kbase + 16 * kti + r, D_, 32 * dc, hf);

    v8f s[2][2];                                 // [key-subtile][q-subtile]
    #pragma unroll
    for (int kti = 0; kti < 2; ++kti)
      #pragma unroll
      for (int q2 = 0; q2 < 2; ++q2) {
        v8f t = {};
        t = wmma_f16(ka[kti][0], qf[q2][0], t);
        t = wmma_f16(ka[kti][1], qf[q2][1], t);
        s[kti][q2] = t;
      }

    v16h vf[4];                                  // V fragments, shared by both q-tiles
    #pragma unroll
    for (int ni = 0; ni < 4; ++ni)
      vf[ni] = load_frag(Vt, 16 * ni + r, T_, kbase, hf);

    #pragma unroll
    for (int q2 = 0; q2 < 2; ++q2) {
      const int qg = qbase + 16 * q2 + r;        // this lane's query for tile q2
      float p0[8], p1[8];
      float bm = -3.0e38f;
      #pragma unroll
      for (int j = 0; j < 8; ++j) {
        int kt0 = kbase + j + 8 * hf;            // S' D-layout: m = j + 8*half (key)
        int kt1 = kt0 + 16;
        float sv0 = (kt0 <= qg) ? (float)s[0][q2][j] * scale : -3.0e38f;
        float sv1 = (kt1 <= qg) ? (float)s[1][q2][j] * scale : -3.0e38f;
        p0[j] = sv0; p1[j] = sv1;
        bm = fmaxf(bm, fmaxf(sv0, sv1));
      }
      bm = fmaxf(bm, __shfl_xor(bm, 16));
      float newm = fmaxf(m_run[q2], bm);
      float corr = __expf(m_run[q2] - newm);
      float rs = 0.0f;
      #pragma unroll
      for (int j = 0; j < 8; ++j) {
        p0[j] = __expf(p0[j] - newm);
        p1[j] = __expf(p1[j] - newm);
        rs += p0[j] + p1[j];
      }
      rs += __shfl_xor(rs, 16);
      l_run[q2] = l_run[q2] * corr + rs;
      m_run[q2] = newm;

      // P in A-frag layout for 16x16x32: i<8 from s-tile0, i>=8 from s-tile1
      v16h pa;
      #pragma unroll
      for (int j = 0; j < 8; ++j) {
        pa[j]     = (_Float16)p0[j];
        pa[j + 8] = (_Float16)p1[j];
      }

      // broadcast row-correction to O accumulator layout (row = j + 8*half)
      float cf[8];
      #pragma unroll
      for (int j = 0; j < 8; ++j) cf[j] = __shfl(corr, j + 8 * hf);

      #pragma unroll
      for (int ni = 0; ni < 4; ++ni) {
        #pragma unroll
        for (int j = 0; j < 8; ++j) o[q2][ni][j] *= cf[j];
        o[q2][ni] = wmma_f16(pa, vf[ni], o[q2][ni]);
      }
    }
  }

  // normalize and store as [B,T,C] f16 for the output projection
  #pragma unroll
  for (int q2 = 0; q2 < 2; ++q2) {
    float linv = 1.0f / l_run[q2];
    float lf[8];
    #pragma unroll
    for (int j = 0; j < 8; ++j) lf[j] = __shfl(linv, j + 8 * hf);
    #pragma unroll
    for (int ni = 0; ni < 4; ++ni)
      #pragma unroll
      for (int j = 0; j < 8; ++j) {
        int qg = qbase + 16 * q2 + j + 8 * hf;
        int d  = 16 * ni + r;
        attnout[((size_t)(b * T_ + qg)) * C_ + h * D_ + d] =
            (_Float16)(o[q2][ni][j] * lf[j]);
      }
  }
}

// ---------------------------------------------------------------------------
// GEMM3: out = attn[4096,1024] @ Wproj + bias  (Wt stored [1024,1024]) -> f32
// ---------------------------------------------------------------------------
__global__ __launch_bounds__(128) void k_gemm_proj(const _Float16* __restrict__ ah,
                                                   const _Float16* __restrict__ wt,
                                                   const float* __restrict__ bias,
                                                   float* __restrict__ out) {
  const int lane = threadIdx.x & 31, wave = threadIdx.x >> 5;
  const int r = lane & 15, hf = lane >> 4;
  const int mbase = blockIdx.y * 128 + (wave >> 1) * 64;
  const int nbase = blockIdx.x * 128 + (wave & 1) * 64;

  v8f acc[4][4] = {};
  gemm_tile64(ah, wt, mbase, nbase, acc, r, hf);

  #pragma unroll
  for (int mi = 0; mi < 4; ++mi)
    #pragma unroll
    for (int ni = 0; ni < 4; ++ni)
      #pragma unroll
      for (int j = 0; j < 8; ++j) {
        int m = mbase + 16 * mi + j + 8 * hf;
        int n = nbase + 16 * ni + r;
        out[(size_t)m * C_ + n] = acc[mi][ni][j] + bias[n];
      }
}

// ---------------------------------------------------------------------------
extern "C" void kernel_launch(void* const* d_in, const int* in_sizes, int n_in,
                              void* d_out, int out_size, void* d_ws, size_t ws_size,
                              hipStream_t stream) {
  const float* x     = (const float*)d_in[0];
  // d_in[1] = causal mask: causality is implemented analytically, not read
  const float* Wqkv  = (const float*)d_in[2];
  const float* Wproj = (const float*)d_in[3];
  const float* bias  = (const float*)d_in[4];
  float* out = (float*)d_out;

  char* ws = (char*)d_ws;
  size_t off = 0;
  auto take = [&](size_t bytes) { char* p = ws + off; off += (bytes + 255) & ~(size_t)255; return p; };

  _Float16* xh    = (_Float16*)take((size_t)BT_ * C_ * 2);   // 8 MiB
  _Float16* wqt   = (_Float16*)take((size_t)N3_ * C_ * 2);   // 6 MiB
  _Float16* wpt   = (_Float16*)take((size_t)C_  * C_ * 2);   // 2 MiB
  _Float16* qb    = (_Float16*)take((size_t)BT_ * C_ * 2);   // 8 MiB  Q [B,H,T,D]
  _Float16* kb    = (_Float16*)take((size_t)BT_ * C_ * 2);   // 8 MiB  K [B,H,T,D]
  _Float16* vtb   = (_Float16*)take((size_t)BT_ * C_ * 2);   // 8 MiB  Vt [B,H,D,T]
  _Float16* attnb = (_Float16*)take((size_t)BT_ * C_ * 2);   // 8 MiB  attn [B,T,C]

  const int nx = BT_ * C_;            // 4194304
  k_f32_to_f16   <<<(nx + 255) / 256, 256, 0, stream>>>(x, xh, nx);
  k_transpose_f16<<<((C_ * N3_) + 255) / 256, 256, 0, stream>>>(Wqkv,  wqt, C_, N3_);
  k_transpose_f16<<<((C_ * C_)  + 255) / 256, 256, 0, stream>>>(Wproj, wpt, C_, C_);

  k_gemm_qkv <<<dim3(N3_ / 128, BT_ / 128), 128, 0, stream>>>(xh, wqt, qb, kb, vtb);
  k_attn     <<<(B_ * H_ * (T_ / 32)) / 8, 256, 0, stream>>>(qb, kb, vtb, attnb);
  k_gemm_proj<<<dim3(C_ / 128, BT_ / 128), 128, 0, stream>>>(attnb, wpt, bias, out);
}
